// CausalSelfAttention_60249801228457
// MI455X (gfx1250) — matrix operand order repaired and checked by reading, not compiled, not run on hardware
//
#include <hip/hip_runtime.h>
#include <hip/hip_bf16.h>

// Problem constants
#define Bn 4
#define Tn 2048
#define Cn 1024
#define Hn 16
#define Dn 64
// SCALE = 1/sqrt(64)
#define QSCALE 0.125f

typedef __attribute__((ext_vector_type(16))) __bf16          v16bf;
typedef __attribute__((ext_vector_type(8)))  float           v8f;
typedef __attribute__((ext_vector_type(8)))  unsigned short  u16x8;

union BF16x16 {
    v16bf v;
    unsigned short u[16];
    u16x8 u8[2];
};

__device__ __forceinline__ v8f vzero8() {
    v8f z = {0.f, 0.f, 0.f, 0.f, 0.f, 0.f, 0.f, 0.f};
    return z;
}

// fp32 -> bf16, round-to-nearest-even
__device__ __forceinline__ unsigned short f2bf(float f) {
    union { float f; unsigned u; } x;
    x.f = f;
    unsigned r = x.u + 0x7FFFu + ((x.u >> 16) & 1u);
    return (unsigned short)(r >> 16);
}

__device__ __forceinline__ v8f wmma_bf16(v16bf a, v16bf b, v8f c) {
    // D = A(16x32) * B(32x16) + C, f32 accumulate
    return __builtin_amdgcn_wmma_f32_16x16x32_bf16(
        false, a, false, b, (short)0, c, false, false);
}

// ---------------------------------------------------------------------------
// Transpose + cast fp32 weight [K,N] row-major -> bf16 [N,K] row-major
// ---------------------------------------------------------------------------
__global__ void tcast_kernel(const float* __restrict__ W,
                             unsigned short* __restrict__ Wt,
                             int K, int N) {
    int i = blockIdx.x * 256 + threadIdx.x;
    if (i >= K * N) return;
    int n = i / K;
    int k = i - n * K;
    Wt[i] = f2bf(W[(size_t)k * N + n]);
}

// ---------------------------------------------------------------------------
// Straight cast fp32 -> bf16 (same layout), vectorized 4 elements/thread
// ---------------------------------------------------------------------------
__global__ void xcast_kernel(const float* __restrict__ X,
                             unsigned short* __restrict__ Xb, int total4) {
    int i = blockIdx.x * 256 + threadIdx.x;
    if (i >= total4) return;
    const float4 v = ((const float4*)X)[i];
    u16x8 dummy; (void)dummy;
    unsigned short o0 = f2bf(v.x), o1 = f2bf(v.y), o2 = f2bf(v.z), o3 = f2bf(v.w);
    typedef __attribute__((ext_vector_type(4))) unsigned short u16x4;
    u16x4 o = {o0, o1, o2, o3};
    ((u16x4*)Xb)[i] = o;
}

// ---------------------------------------------------------------------------
// QKV GEMM:  qkv = Xb[8192,1024](bf16) @ Wt^T + b,  Wt bf16 [3072,1024] (N,K)
// Writes q (scaled), k in [B,H,T,D] bf16; v transposed into [B,H,D,T] bf16.
// Block = 256 threads = 8 waves (2 M x 4 N), wave tile 64x64.
// Grid = (3072/256, 8192/128) = (12, 64)
// ---------------------------------------------------------------------------
__global__ __launch_bounds__(256) void qkv_gemm_kernel(
    const unsigned short* __restrict__ A, const unsigned short* __restrict__ Wt,
    const float* __restrict__ bias, unsigned short* __restrict__ qb,
    unsigned short* __restrict__ kb, unsigned short* __restrict__ vtb)
{
    const int K = Cn;
    const int lane = threadIdx.x & 31;
    const int w    = threadIdx.x >> 5;
    const int wm = w >> 2, wn = w & 3;
    const int m0 = blockIdx.y * 128 + wm * 64;
    const int n0 = blockIdx.x * 256 + wn * 64;
    const int lr = lane & 15;
    const int half = lane >> 4;

    v8f acc[4][4];
#pragma unroll
    for (int mt = 0; mt < 4; ++mt)
#pragma unroll
        for (int nt = 0; nt < 4; ++nt) acc[mt][nt] = vzero8();

#pragma unroll 2
    for (int k0 = 0; k0 < K; k0 += 32) {
        BF16x16 af[4];
#pragma unroll
        for (int mt = 0; mt < 4; ++mt) {
            const unsigned short* ap = A + (size_t)(m0 + mt * 16 + lr) * K + k0;
            af[mt].u8[0] = *(const u16x8*)(ap + half * 16);
            af[mt].u8[1] = *(const u16x8*)(ap + half * 16 + 8);
        }
        BF16x16 bf[4];
#pragma unroll
        for (int nt = 0; nt < 4; ++nt) {
            const unsigned short* bp =
                Wt + (size_t)(n0 + nt * 16 + lr) * K + k0 + half * 16;
            bf[nt].u8[0] = *(const u16x8*)bp;
            bf[nt].u8[1] = *(const u16x8*)(bp + 8);
        }
#pragma unroll
        for (int mt = 0; mt < 4; ++mt)
#pragma unroll
            for (int nt = 0; nt < 4; ++nt)
                acc[mt][nt] = wmma_bf16(af[mt].v, bf[nt].v, acc[mt][nt]);
    }

    // Epilogue: bias + scatter into q/k/vT layouts
#pragma unroll
    for (int nt = 0; nt < 4; ++nt) {
        const int n = n0 + nt * 16 + lr;
        const float bn = bias[n];
        const int part = n >> 10;
        const int rem  = n & 1023;
        const int h = rem >> 6;
        const int d = rem & 63;
#pragma unroll
        for (int mt = 0; mt < 4; ++mt) {
#pragma unroll
            for (int r = 0; r < 8; ++r) {
                const int m = m0 + mt * 16 + r + half * 8;
                const int b = m >> 11;
                const int t = m & 2047;
                float val = acc[mt][nt][r] + bn;
                if (part == 0) {
                    qb[(((size_t)(b * Hn + h)) * Tn + t) * Dn + d] =
                        f2bf(val * QSCALE);
                } else if (part == 1) {
                    kb[(((size_t)(b * Hn + h)) * Tn + t) * Dn + d] = f2bf(val);
                } else {
                    vtb[(((size_t)(b * Hn + h)) * Dn + d) * Tn + t] = f2bf(val);
                }
            }
        }
    }
}

// ---------------------------------------------------------------------------
// Flash attention.  Per wave: one 64-query tile of one (b,h).
// S^T = K*Q^T so the softmaxed tile is directly the A-fragment for P*V.
// Grid = (B*H, T/(64*8)) = (64, 4), block = 256 (8 waves).
// ---------------------------------------------------------------------------
__global__ __launch_bounds__(256) void attn_kernel(
    const unsigned short* __restrict__ qb, const unsigned short* __restrict__ kb,
    const unsigned short* __restrict__ vtb, unsigned short* __restrict__ ob)
{
    const int lane = threadIdx.x & 31;
    const int w    = threadIdx.x >> 5;
    const int lr   = lane & 15;
    const int half = lane >> 4;
    const int bh = blockIdx.x;
    const int b = bh >> 4, h = bh & 15;
    const int m0 = (blockIdx.y * 8 + w) * 64;
    const size_t base = (size_t)bh * Tn * Dn;   // same size for q,k and vT

    // Q^T B-fragments (held for whole loop): lane col = query, elems = d
    BF16x16 qf[4][2];
#pragma unroll
    for (int j = 0; j < 4; ++j)
#pragma unroll
        for (int ks = 0; ks < 2; ++ks) {
            const unsigned short* qp =
                qb + base + (size_t)(m0 + j * 16 + lr) * Dn + ks * 32 + half * 16;
            qf[j][ks].u8[0] = *(const u16x8*)qp;
            qf[j][ks].u8[1] = *(const u16x8*)(qp + 8);
        }

    v8f oacc[4][4];          // [q subtile][d subtile], rows=q, cols=d
#pragma unroll
    for (int j = 0; j < 4; ++j)
#pragma unroll
        for (int dt = 0; dt < 4; ++dt) oacc[j][dt] = vzero8();

    float mrun[4], lrun[4];
#pragma unroll
    for (int j = 0; j < 4; ++j) { mrun[j] = -1e30f; lrun[j] = 0.f; }

    const int send = m0 + 64;
#pragma unroll 1
    for (int s0 = 0; s0 < send; s0 += 32) {
        // K A-fragments: lane row = key, elems = d
        BF16x16 kf[2][2];
#pragma unroll
        for (int st = 0; st < 2; ++st)
#pragma unroll
            for (int ks = 0; ks < 2; ++ks) {
                const unsigned short* kp =
                    kb + base + (size_t)(s0 + st * 16 + lr) * Dn + ks * 32;
                kf[st][ks].u8[0] = *(const u16x8*)(kp + half * 16);
                kf[st][ks].u8[1] = *(const u16x8*)(kp + half * 16 + 8);
            }

        // S^T = K * Q^T  (rows = key, cols = query)
        v8f sacc[2][4];
#pragma unroll
        for (int st = 0; st < 2; ++st)
#pragma unroll
            for (int j = 0; j < 4; ++j) sacc[st][j] = vzero8();
#pragma unroll
        for (int ks = 0; ks < 2; ++ks)
#pragma unroll
            for (int st = 0; st < 2; ++st)
#pragma unroll
                for (int j = 0; j < 4; ++j)
                    sacc[st][j] = wmma_bf16(kf[st][ks].v, qf[j][ks].v, sacc[st][j]);

        // V B-fragments: lane col = d, elems = key (contiguous thanks to vT)
        BF16x16 vf[4];
#pragma unroll
        for (int dt = 0; dt < 4; ++dt) {
            const unsigned short* vp =
                vtb + base + (size_t)(dt * 16 + lr) * Tn + s0 + half * 8;
            vf[dt].u8[0] = *(const u16x8*)vp;
            vf[dt].u8[1] = *(const u16x8*)(vp + 16);
        }

        // Online softmax per query subtile + P*V accumulate
#pragma unroll
        for (int j = 0; j < 4; ++j) {
            const int q = m0 + j * 16 + lr;
            float vals[2][8];
            float mt = -1e30f;
#pragma unroll
            for (int st = 0; st < 2; ++st)
#pragma unroll
                for (int r = 0; r < 8; ++r) {
                    const int s = s0 + st * 16 + r + half * 8;
                    float v = sacc[st][j][r];
                    if (s > q) v = -1e30f;      // causal mask
                    vals[st][r] = v;
                    mt = fmaxf(mt, v);
                }
            mt = fmaxf(mt, __shfl_xor(mt, 16));
            const float mn = fmaxf(mrun[j], mt);
            const float f  = __expf(mrun[j] - mn);
            mrun[j] = mn;

            BF16x16 pf;          // A-fragment of P, no cross-lane movement
            float rs = 0.f;
#pragma unroll
            for (int st = 0; st < 2; ++st)
#pragma unroll
                for (int r = 0; r < 8; ++r) {
                    const float p = __expf(vals[st][r] - mn);
                    rs += p;
                    pf.u[st * 8 + r] = f2bf(p);
                }
            rs += __shfl_xor(rs, 16);
            lrun[j] = lrun[j] * f + rs;

            // rescale O rows (row q' = r + 8*half lives in lane q'%16)
#pragma unroll
            for (int r = 0; r < 8; ++r) {
                const float fb = __shfl(f, r + half * 8);
#pragma unroll
                for (int dt = 0; dt < 4; ++dt) oacc[j][dt][r] *= fb;
            }
#pragma unroll
            for (int dt = 0; dt < 4; ++dt)
                oacc[j][dt] = wmma_bf16(pf.v, vf[dt].v, oacc[j][dt]);
        }
    }

    // Normalize and store bf16 attention output [B,T,C]
#pragma unroll
    for (int j = 0; j < 4; ++j) {
        const float inv = 1.0f / lrun[j];
#pragma unroll
        for (int r = 0; r < 8; ++r) {
            const float il = __shfl(inv, r + half * 8);
            const int t = m0 + j * 16 + r + half * 8;
            const size_t rowbase = ((size_t)b * Tn + t) * Cn + h * Dn;
#pragma unroll
            for (int dt = 0; dt < 4; ++dt)
                ob[rowbase + dt * 16 + lr] = f2bf(oacc[j][dt][r] * il);
        }
    }
}

// ---------------------------------------------------------------------------
// Output projection: out[8192,1024] = A_bf16 @ Wt^T + bias (fp32 out)
// Grid = (1024/256, 8192/128) = (4, 64)
// ---------------------------------------------------------------------------
__global__ __launch_bounds__(256) void out_gemm_kernel(
    const unsigned short* __restrict__ A, const unsigned short* __restrict__ Wt,
    const float* __restrict__ bias, float* __restrict__ out)
{
    const int K = Cn;
    const int lane = threadIdx.x & 31;
    const int w    = threadIdx.x >> 5;
    const int wm = w >> 2, wn = w & 3;
    const int m0 = blockIdx.y * 128 + wm * 64;
    const int n0 = blockIdx.x * 256 + wn * 64;
    const int lr = lane & 15;
    const int half = lane >> 4;

    v8f acc[4][4];
#pragma unroll
    for (int mt = 0; mt < 4; ++mt)
#pragma unroll
        for (int nt = 0; nt < 4; ++nt) acc[mt][nt] = vzero8();

#pragma unroll 2
    for (int k0 = 0; k0 < K; k0 += 32) {
        BF16x16 af[4];
#pragma unroll
        for (int mt = 0; mt < 4; ++mt) {
            const unsigned short* ap = A + (size_t)(m0 + mt * 16 + lr) * K + k0;
            af[mt].u8[0] = *(const u16x8*)(ap + half * 16);
            af[mt].u8[1] = *(const u16x8*)(ap + half * 16 + 8);
        }
        BF16x16 bf[4];
#pragma unroll
        for (int nt = 0; nt < 4; ++nt) {
            const unsigned short* bp =
                Wt + (size_t)(n0 + nt * 16 + lr) * K + k0 + half * 16;
            bf[nt].u8[0] = *(const u16x8*)bp;
            bf[nt].u8[1] = *(const u16x8*)(bp + 8);
        }
#pragma unroll
        for (int mt = 0; mt < 4; ++mt)
#pragma unroll
            for (int nt = 0; nt < 4; ++nt)
                acc[mt][nt] = wmma_bf16(af[mt].v, bf[nt].v, acc[mt][nt]);
    }

#pragma unroll
    for (int nt = 0; nt < 4; ++nt) {
        const int n = n0 + nt * 16 + lr;
        const float bn = bias[n];
#pragma unroll
        for (int mt = 0; mt < 4; ++mt)
#pragma unroll
            for (int r = 0; r < 8; ++r) {
                const int m = m0 + mt * 16 + r + half * 8;
                out[(size_t)m * Cn + n] = acc[mt][nt][r] + bn;
            }
    }
}

// ---------------------------------------------------------------------------
extern "C" void kernel_launch(void* const* d_in, const int* in_sizes, int n_in,
                              void* d_out, int out_size, void* d_ws, size_t ws_size,
                              hipStream_t stream) {
    const float* x    = (const float*)d_in[0];
    const float* Wqkv = (const float*)d_in[1];
    const float* bqkv = (const float*)d_in[2];
    const float* Wout = (const float*)d_in[3];
    const float* bout = (const float*)d_in[4];
    float* out = (float*)d_out;

    const size_t NQKV = (size_t)3 * Cn * Cn;        // 3,145,728
    const size_t NOUT = (size_t)Cn * Cn;            // 1,048,576
    const size_t NHTD = (size_t)Bn * Hn * Tn * Dn;  // 8,388,608
    const size_t NX   = (size_t)Bn * Tn * Cn;       // 8,388,608

    unsigned short* wqkvt = (unsigned short*)d_ws;
    unsigned short* woutt = wqkvt + NQKV;
    unsigned short* qb    = woutt + NOUT;
    unsigned short* kb    = qb + NHTD;
    unsigned short* vtb   = kb + NHTD;
    unsigned short* ob    = vtb + NHTD;
    unsigned short* xb    = ob + NX;

    tcast_kernel<<<(int)((NQKV + 255) / 256), 256, 0, stream>>>(Wqkv, wqkvt, Cn, 3 * Cn);
    tcast_kernel<<<(int)((NOUT + 255) / 256), 256, 0, stream>>>(Wout, woutt, Cn, Cn);
    xcast_kernel<<<(int)((NX / 4 + 255) / 256), 256, 0, stream>>>(x, xb, (int)(NX / 4));

    qkv_gemm_kernel<<<dim3(12, 64), 256, 0, stream>>>(xb, wqkvt, bqkv, qb, kb, vtb);
    attn_kernel<<<dim3(Bn * Hn, Tn / (64 * 8)), 256, 0, stream>>>(qb, kb, vtb, ob);
    out_gemm_kernel<<<dim3(4, 64), 256, 0, stream>>>(ob, woutt, bout, out);
}
